// GraphSAGENet_57200374448341
// MI455X (gfx1250) — compile-verified, matrix operand order checked by reading, not run on hardware
//
#include <hip/hip_runtime.h>

#define N_NODES 50000
#define N_EDGES 250000
#define D_IN    768
#define D_HID   256
#define D_OUT   2

typedef __attribute__((ext_vector_type(16))) __bf16 v16bf;
typedef __attribute__((ext_vector_type(8)))  __bf16 v8bf;
typedef __attribute__((ext_vector_type(4)))  __bf16 v4bf;
typedef __attribute__((ext_vector_type(8)))  float  v8f;
typedef __attribute__((ext_vector_type(4)))  float  v4f;

// ---------------------------------------------------------------- utilities
__global__ void zero_f32(float* __restrict__ p, int n) {
  int i = blockIdx.x * blockDim.x + threadIdx.x;
  if (i < n) p[i] = 0.0f;
}

// Concatenate W1_l (256x768) and W1_r (256x768) into one bf16 [512,768] buffer.
__global__ void cvt_w1(const float* __restrict__ wl, const float* __restrict__ wr,
                       __bf16* __restrict__ wcat) {
  int i = blockIdx.x * blockDim.x + threadIdx.x;
  const int half = D_HID * D_IN;           // 196608
  if (i < half)            wcat[i] = (__bf16)wl[i];
  else if (i < 2 * half)   wcat[i] = (__bf16)wr[i - half];
}

// ------------------------------------------------------- layer-1 dual GEMM
// XWl = x @ W1_l.T ; XWr = x @ W1_r.T   (both [50000,256], f32 accum)
// Block: 256 threads = 8 waves. Block owns a 16-node row tile; wave w owns
// 4 column tiles of the concatenated 512-wide output. K loop: 24 x 32.
__global__ void __launch_bounds__(256)
gemm1(const float* __restrict__ x, const __bf16* __restrict__ wcat,
      float* __restrict__ xwl, float* __restrict__ xwr) {
  const int lane = threadIdx.x & 31;
  const int wave = threadIdx.x >> 5;
  const int col  = lane & 15;     // A row index / B col index within tile
  const int hi   = lane >> 4;     // lane-half selects K sub-pattern
  const int koff = hi * 8;
  const int nodeBase = blockIdx.x * 16;

  v8f acc[4] = {};

  const float* xrow = x + (size_t)(nodeBase + col) * D_IN + koff;

  for (int kb = 0; kb < D_IN; kb += 32) {
    // ---- A fragment: row (col), K = kb+koff..+7 and kb+16+koff..+7 (f32 -> bf16)
    v4f a0 = *(const v4f*)(xrow + kb);
    v4f a1 = *(const v4f*)(xrow + kb + 4);
    v4f a2 = *(const v4f*)(xrow + kb + 16);
    v4f a3 = *(const v4f*)(xrow + kb + 20);
    v4bf c0 = __builtin_convertvector(a0, v4bf);
    v4bf c1 = __builtin_convertvector(a1, v4bf);
    v4bf c2 = __builtin_convertvector(a2, v4bf);
    v4bf c3 = __builtin_convertvector(a3, v4bf);
    v8bf lo = __builtin_shufflevector(c0, c1, 0, 1, 2, 3, 4, 5, 6, 7);
    v8bf hh = __builtin_shufflevector(c2, c3, 0, 1, 2, 3, 4, 5, 6, 7);
    v16bf afrag = __builtin_shufflevector(lo, hh,
        0, 1, 2, 3, 4, 5, 6, 7, 8, 9, 10, 11, 12, 13, 14, 15);

#pragma unroll
    for (int j = 0; j < 4; ++j) {
      const int gt = wave * 4 + j;                       // 0..31 column tile
      const __bf16* wrow = wcat + (size_t)(gt * 16 + col) * D_IN + kb + koff;
      v8bf b0 = *(const v8bf*)(wrow);
      v8bf b1 = *(const v8bf*)(wrow + 16);
      v16bf bfrag = __builtin_shufflevector(b0, b1,
          0, 1, 2, 3, 4, 5, 6, 7, 8, 9, 10, 11, 12, 13, 14, 15);
      acc[j] = __builtin_amdgcn_wmma_f32_16x16x32_bf16(
          false, afrag, false, bfrag, (short)0, acc[j], false, false);
    }
  }

  // ---- store C: VGPR i -> row (i + hi*8), col = lane&15 within tile
#pragma unroll
  for (int j = 0; j < 4; ++j) {
    const int gt = wave * 4 + j;
    float* outp = (gt < 16) ? xwl : xwr;
    const int colBase = (gt & 15) * 16;
#pragma unroll
    for (int i = 0; i < 8; ++i) {
      const int row = nodeBase + hi * 8 + i;
      outp[(size_t)row * D_HID + colBase + col] = acc[j][i];
    }
  }
}

// ------------------------------------------------- layer-1 edge aggregation
// 64 threads per edge, each handles 4 consecutive features (float4 gather,
// 4 global f32 atomic adds). Chunk 0 also counts in-degree.
__global__ void aggregate1(const long long* __restrict__ srcI,
                           const long long* __restrict__ dstI,
                           const float* __restrict__ xwl,
                           float* __restrict__ agg1, float* __restrict__ cnt) {
  const int idx = blockIdx.x * blockDim.x + threadIdx.x;  // < 16,000,000
  const int e = idx >> 6;
  const int c = (idx & 63) << 2;
  const int s = (int)srcI[e];
  const int d = (int)dstI[e];
  const float4 v = *(const float4*)(xwl + (size_t)s * D_HID + c);
  float* ap = agg1 + (size_t)d * D_HID + c;
  atomicAdd(ap + 0, v.x);
  atomicAdd(ap + 1, v.y);
  atomicAdd(ap + 2, v.z);
  atomicAdd(ap + 3, v.w);
  if (c == 0) atomicAdd(cnt + d, 1.0f);
}

// h = relu(agg/max(cnt,1) + b1 + XWr) stored as bf16
__global__ void finalize1(const float* __restrict__ agg1,
                          const float* __restrict__ cnt,
                          const float* __restrict__ b1,
                          const float* __restrict__ xwr,
                          __bf16* __restrict__ hbf) {
  const int i = blockIdx.x * blockDim.x + threadIdx.x;   // < 12,800,000
  if (i >= N_NODES * D_HID) return;
  const int n = i >> 8;
  const int c = i & 255;
  float cn = cnt[n];
  float denom = cn > 1.0f ? cn : 1.0f;
  float v = agg1[i] / denom + b1[c] + xwr[i];
  v = v > 0.0f ? v : 0.0f;
  hbf[i] = (__bf16)v;
}

// --------------------------------------------------------- layer-2 (256->2)
// Wave per node: lane-parallel dot products + shfl_xor reduction.
__global__ void __launch_bounds__(256)
gemm2(const __bf16* __restrict__ hbf, const float* __restrict__ w2l,
      const float* __restrict__ w2r, float* __restrict__ hl,
      float* __restrict__ hr) {
  const int lane = threadIdx.x & 31;
  const int wave = threadIdx.x >> 5;
  const int node = blockIdx.x * 8 + wave;
  const __bf16* hrow = hbf + (size_t)node * D_HID;
  float al0 = 0.f, al1 = 0.f, ar0 = 0.f, ar1 = 0.f;
  for (int k = lane; k < D_HID; k += 32) {
    const float hv = (float)hrow[k];
    al0 += hv * w2l[k];
    al1 += hv * w2l[D_HID + k];
    ar0 += hv * w2r[k];
    ar1 += hv * w2r[D_HID + k];
  }
#pragma unroll
  for (int m = 16; m >= 1; m >>= 1) {
    al0 += __shfl_xor(al0, m, 32);
    al1 += __shfl_xor(al1, m, 32);
    ar0 += __shfl_xor(ar0, m, 32);
    ar1 += __shfl_xor(ar1, m, 32);
  }
  if (lane == 0) {
    hl[node * 2 + 0] = al0;
    hl[node * 2 + 1] = al1;
    hr[node * 2 + 0] = ar0;
    hr[node * 2 + 1] = ar1;
  }
}

__global__ void aggregate2(const long long* __restrict__ srcI,
                           const long long* __restrict__ dstI,
                           const float* __restrict__ hl,
                           float* __restrict__ agg2) {
  const int e = blockIdx.x * blockDim.x + threadIdx.x;
  if (e >= N_EDGES) return;
  const int s = (int)srcI[e];
  const int d = (int)dstI[e];
  atomicAdd(agg2 + d * 2 + 0, hl[s * 2 + 0]);
  atomicAdd(agg2 + d * 2 + 1, hl[s * 2 + 1]);
}

__global__ void finalize2(const float* __restrict__ agg2,
                          const float* __restrict__ cnt,
                          const float* __restrict__ b2,
                          const float* __restrict__ hr,
                          float* __restrict__ out) {
  const int i = blockIdx.x * blockDim.x + threadIdx.x;   // < 100,000
  if (i >= N_NODES * D_OUT) return;
  const int n = i >> 1;
  const int c = i & 1;
  float cn = cnt[n];
  float denom = cn > 1.0f ? cn : 1.0f;
  out[i] = agg2[i] / denom + b2[c] + hr[i];
}

// ------------------------------------------------------------------ driver
extern "C" void kernel_launch(void* const* d_in, const int* in_sizes, int n_in,
                              void* d_out, int out_size, void* d_ws, size_t ws_size,
                              hipStream_t stream) {
  (void)in_sizes; (void)n_in; (void)out_size; (void)ws_size;
  const float*     x    = (const float*)d_in[0];
  const long long* ei   = (const long long*)d_in[1];
  const float*     w1l  = (const float*)d_in[2];
  const float*     b1   = (const float*)d_in[3];
  const float*     w1r  = (const float*)d_in[4];
  const float*     w2l  = (const float*)d_in[5];
  const float*     b2   = (const float*)d_in[6];
  const float*     w2r  = (const float*)d_in[7];
  float*           out  = (float*)d_out;

  char* ws = (char*)d_ws;
  size_t off = 0;
  auto take = [&](size_t bytes) -> char* {
    char* p = ws + off;
    off += (bytes + 255) & ~(size_t)255;
    return p;
  };
  float*  agg1 = (float*) take((size_t)N_NODES * D_HID * 4);
  float*  cnt  = (float*) take((size_t)N_NODES * 4);
  float*  agg2 = (float*) take((size_t)N_NODES * D_OUT * 4);
  __bf16* wcat = (__bf16*)take((size_t)2 * D_HID * D_IN * 2);
  float*  xwl  = (float*) take((size_t)N_NODES * D_HID * 4);
  float*  xwr  = (float*) take((size_t)N_NODES * D_HID * 4);
  __bf16* hbf  = (__bf16*)take((size_t)N_NODES * D_HID * 2);
  float*  hl   = (float*) take((size_t)N_NODES * D_OUT * 4);
  float*  hr   = (float*) take((size_t)N_NODES * D_OUT * 4);

  const long long* srcI = ei;              // edge_index[0]
  const long long* dstI = ei + N_EDGES;    // edge_index[1]

  // re-zero atomic accumulators every call (graph replays must be deterministic)
  zero_f32<<<(N_NODES * D_HID + 255) / 256, 256, 0, stream>>>(agg1, N_NODES * D_HID);
  zero_f32<<<(N_NODES + 255) / 256, 256, 0, stream>>>(cnt, N_NODES);
  zero_f32<<<(N_NODES * D_OUT + 255) / 256, 256, 0, stream>>>(agg2, N_NODES * D_OUT);

  cvt_w1<<<(2 * D_HID * D_IN + 255) / 256, 256, 0, stream>>>(w1l, w1r, wcat);

  gemm1<<<N_NODES / 16, 256, 0, stream>>>(x, wcat, xwl, xwr);

  aggregate1<<<(N_EDGES * 64) / 256, 256, 0, stream>>>(srcI, dstI, xwl, agg1, cnt);

  finalize1<<<N_NODES, 256, 0, stream>>>(agg1, cnt, b1, xwr, hbf);

  gemm2<<<N_NODES / 8, 256, 0, stream>>>(hbf, w2l, w2r, hl, hr);

  aggregate2<<<(N_EDGES + 255) / 256, 256, 0, stream>>>(srcI, dstI, hl, agg2);

  finalize2<<<(N_NODES * D_OUT + 255) / 256, 256, 0, stream>>>(agg2, cnt, b2, hr, out);
}
